// Attention_74182675136543
// MI455X (gfx1250) — compile-verified, matrix operand order checked by reading, not run on hardware
//
#include <hip/hip_runtime.h>

// Problem constants (match reference)
#define HIDN  2048
#define NH    32
#define NKVH  8
#define HD    64
#define BB    2
#define SSQ   2048
#define MROWS (BB*SSQ)   // 4096 flattened rows of x / attn-out

typedef __attribute__((ext_vector_type(16))) __bf16 v16bf;
typedef __attribute__((ext_vector_type(8)))  float  v8f;

__device__ __forceinline__ v8f wmma_bf16(v16bf a, v16bf b, v8f c) {
  // D = A(16x32 bf16) * B(32x16 bf16) + C(16x16 f32)
  return __builtin_amdgcn_wmma_f32_16x16x32_bf16(
      /*neg_a=*/false, a, /*neg_b=*/false, b,
      /*c_mod=*/(short)0, c, /*reuse_a=*/false, /*reuse_b=*/false);
}

// Load one 16x32 bf16 WMMA fragment (A-layout; B uses the same mapping with
// row<->column roles swapped). Per ISA 7.12.2: for lane half h, vector elems
// 0..7 hold K = 8h..8h+7 and elems 8..15 hold K = 16+8h..23+8h -> two 16B loads.
__device__ __forceinline__ v16bf load_frag16(const __bf16* base, int ld,
                                             int row, int half, int kbase) {
  union U { uint4 c[2]; v16bf v; } f;
  const __bf16* p = base + (size_t)row * ld + kbase + 8 * half;
  f.c[0] = *(const uint4*)(p);
  f.c[1] = *(const uint4*)(p + 16);
  return f.v;
}

// CDNA5 async copy: 16B per lane, global -> LDS, tracked by ASYNCcnt.
__device__ __forceinline__ void async_copy16(unsigned lds_byte_off, const void* gaddr) {
  asm volatile("global_load_async_to_lds_b128 %0, %1, off"
               :: "v"(lds_byte_off), "v"(gaddr) : "memory");
}

// ---------------- conversion / transpose kernels ----------------

__global__ void k_f32_to_bf16(const float* __restrict__ src,
                              unsigned short* __restrict__ dstr, int n) {
  __bf16* dst = (__bf16*)dstr;
  int i = blockIdx.x * blockDim.x + threadIdx.x;
  if (i < n) dst[i] = (__bf16)src[i];
}

// W: K x N f32 (row-major)  ->  Wt: N x K bf16 (row-major)
__global__ void k_transpose_bf16(const float* __restrict__ W,
                                 unsigned short* __restrict__ Wtr,
                                 int K, int N) {
  __bf16* Wt = (__bf16*)Wtr;
  int i = blockIdx.x * blockDim.x + threadIdx.x;
  if (i >= K * N) return;
  int k = i / N, n = i - k * N;
  Wt[(size_t)n * K + k] = (__bf16)W[(size_t)k * N + n];
}

// ---------------- QKV projection + fused RoPE ----------------
// 4 waves/WG; WG computes 128 rows x one head's 64 cols. B tile (64x32 bf16,
// 4KB) is double-buffered in LDS via async-to-LDS; each wave does a 32x64
// register tile = 8 WMMAs per K-step.
// Q/K stored [b][head][s][d]; V stored transposed [b][head][d][s].
__global__ void __launch_bounds__(128)
k_gemm_qkv(const unsigned short* __restrict__ Xr,
           const unsigned short* __restrict__ Wtr,
           const float* __restrict__ cosT,
           const float* __restrict__ sinT,
           unsigned short* __restrict__ outr,
           int nheads, int do_rope, int v_transposed) {
  const __bf16* X  = (const __bf16*)Xr;
  const __bf16* Wt = (const __bf16*)Wtr;
  __bf16* out = (__bf16*)outr;

  __shared__ __align__(16) __bf16 Bt[2][64 * 32];   // double-buffered B tile

  const int t    = threadIdx.x;
  const int lane = t & 31;
  const int w    = t >> 5;             // wave id 0..3
  const int nl   = lane & 15;
  const int half = lane >> 4;
  const int row0 = blockIdx.x * 128 + w * 32;
  const int head = blockIdx.y;

  const __bf16* Abase = X  + (size_t)row0 * HIDN;
  const __bf16* Bbase = Wt + (size_t)head * HD * HIDN;

  // stage Wt[head*64 .. +63][k0..k0+31] into Bt[buf]; constant-trip loop so
  // no divergent bound check / exec masking around the async ops.
  auto stage = [&](int buf, int k0) {
#pragma unroll
    for (int i = 0; i < 2; ++i) {
      int c  = t + i * 128;             // chunk 0..255
      int n  = c >> 2;                  // output col within head block
      int ke = (c & 3) * 8;             // element offset within 32-wide row
      async_copy16((unsigned)(size_t)&Bt[buf][n * 32 + ke],
                   Bbase + (size_t)n * HIDN + k0 + ke);
    }
  };

  v8f acc[2][4] = {};
  stage(0, 0);
  for (int kt = 0; kt < HIDN / 32; ++kt) {
    const int k0 = kt * 32;
    const bool hasnext = (kt + 1) < (HIDN / 32);
    if (hasnext) {
      stage((kt + 1) & 1, k0 + 32);                      // prefetch next tile
      asm volatile("s_wait_asynccnt 2" ::: "memory");    // current tile done
    } else {
      asm volatile("s_wait_asynccnt 0" ::: "memory");
    }
    __syncthreads();

    const __bf16* Bc = Bt[kt & 1];
    v16bf a0 = load_frag16(Abase,                     HIDN, nl, half, k0);
    v16bf a1 = load_frag16(Abase + (size_t)16 * HIDN, HIDN, nl, half, k0);
#pragma unroll
    for (int f = 0; f < 4; ++f) {
      v16bf b = load_frag16(Bc, 32, f * 16 + nl, half, 0);
      acc[0][f] = wmma_bf16(a0, b, acc[0][f]);
      acc[1][f] = wmma_bf16(a1, b, acc[1][f]);
    }
    __syncthreads();   // all reads of this buffer done before it is re-staged
  }

  // Epilogue: fused RoPE (partner d +/- 32 lives in fragment f +/- 2, same lane/vgpr)
#pragma unroll
  for (int mb = 0; mb < 2; ++mb) {
    float res[4][8];
#pragma unroll
    for (int f = 0; f < 4; ++f) {
#pragma unroll
      for (int v = 0; v < 8; ++v) {
        int grow = row0 + mb * 16 + v + 8 * half;
        int s    = grow & (SSQ - 1);
        int d    = nl + 16 * f;
        float val = acc[mb][f][v];
        if (do_rope) {
          float c  = cosT[s * HD + d];
          float si = sinT[s * HD + d];
          float partner = (f < 2) ? acc[mb][f + 2][v] : acc[mb][f - 2][v];
          val = (f < 2) ? (val * c - partner * si) : (val * c + partner * si);
        }
        res[f][v] = val;
      }
    }
#pragma unroll
    for (int f = 0; f < 4; ++f) {
#pragma unroll
      for (int v = 0; v < 8; ++v) {
        int grow = row0 + mb * 16 + v + 8 * half;
        int s    = grow & (SSQ - 1);
        int b    = grow >> 11;               // grow / SSQ
        int d    = nl + 16 * f;
        size_t idx = v_transposed
            ? ((((size_t)b * nheads + head) * HD + d) * SSQ + s)
            : ((((size_t)b * nheads + head) * SSQ + s) * HD + d);
        out[idx] = (__bf16)res[f][v];
      }
    }
  }
}

// ---------------- flash attention (causal, GQA) ----------------
// 4 waves/WG covering 128 q rows; K tile (32x64) and V tile (64x32, stored
// transposed) double-buffered in LDS via async-to-LDS and shared by all 4
// waves (8x less K/V traffic vs per-16-row-tile loads). Each wave holds two
// 16-row q blocks -> 16 WMMAs per 32-key block. Causal skip is wave-uniform.
__global__ void __launch_bounds__(128)
k_attn(const unsigned short* __restrict__ Qr,
       const unsigned short* __restrict__ Kr,
       const unsigned short* __restrict__ Vtr,
       unsigned short* __restrict__ Or) {
  const __bf16* Q  = (const __bf16*)Qr;
  const __bf16* Kc = (const __bf16*)Kr;
  const __bf16* Vt = (const __bf16*)Vtr;
  __bf16* O = (__bf16*)Or;

  __shared__ __align__(16) __bf16 Ktile[2][32 * 64];   // [key][d], 4KB each
  __shared__ __align__(16) __bf16 Vtile[2][64 * 32];   // [d][key], 4KB each
  __shared__ __align__(16) __bf16 Pld[4][2][16 * 32];  // per-wave P staging

  const int t    = threadIdx.x;
  const int lane = t & 31;
  const int w    = t >> 5;
  const int nl   = lane & 15;
  const int half = lane >> 4;
  const int h    = blockIdx.y;
  const int b    = blockIdx.z;
  const int kvh  = h >> 2;                        // H / KVH = 4
  const int q0   = blockIdx.x * 128 + w * 32;     // this wave's first q row

  const __bf16* Qb = Q  + (((size_t)b * NH   + h  ) * SSQ + q0) * HD;
  const __bf16* Kb = Kc + (((size_t)b * NKVH + kvh) * SSQ) * HD;
  const __bf16* Vb = Vt + (((size_t)b * NKVH + kvh) * HD ) * SSQ;

  // Q A-fragments: 2 q-blocks x full head dim
  v16bf qa[2][2];
#pragma unroll
  for (int mb = 0; mb < 2; ++mb)
#pragma unroll
    for (int kc = 0; kc < 2; ++kc)
      qa[mb][kc] = load_frag16(Qb + (size_t)mb * 16 * HD, HD, nl, half, kc * 32);

  v8f   oacc[2][4] = {};
  float mrow[2][8], lrow[2][8];
#pragma unroll
  for (int mb = 0; mb < 2; ++mb)
#pragma unroll
    for (int v = 0; v < 8; ++v) { mrow[mb][v] = -1e30f; lrow[mb][v] = 0.f; }

  // stage K[kb..kb+31][0..63] and Vt[0..63][kb..kb+31] (4KB each, 4 asyncs/thread)
  auto stage = [&](int buf, int kb) {
#pragma unroll
    for (int i = 0; i < 2; ++i) {
      int c  = t + i * 128;
      int kk = c >> 3, ke = (c & 7) * 8;          // K tile: 32 rows x 8 chunks
      async_copy16((unsigned)(size_t)&Ktile[buf][kk * 64 + ke],
                   Kb + (size_t)(kb + kk) * HD + ke);
    }
#pragma unroll
    for (int i = 0; i < 2; ++i) {
      int c = t + i * 128;
      int d = c >> 2, ke = (c & 3) * 8;           // V tile: 64 rows x 4 chunks
      async_copy16((unsigned)(size_t)&Vtile[buf][d * 32 + ke],
                   Vb + (size_t)d * SSQ + kb + ke);
    }
  };

  const int nkb = blockIdx.x * 4 + 4;             // 32-key blocks (causal bound for WG)
  stage(0, 0);
  for (int it = 0; it < nkb; ++it) {
    const int kb = it * 32;
    const bool hasnext = (it + 1) < nkb;
    if (hasnext) {
      stage((it + 1) & 1, kb + 32);
      asm volatile("s_wait_asynccnt 4" ::: "memory");
    } else {
      asm volatile("s_wait_asynccnt 0" ::: "memory");
    }
    __syncthreads();
    const __bf16* Kl = Ktile[it & 1];
    const __bf16* Vl = Vtile[it & 1];

    if (kb <= q0 + 31) {                          // wave-uniform causal skip
      // ---- scores: S(2x16 x 32) = Q @ K^T ----
      v8f sc[2][2] = {};
#pragma unroll
      for (int nb = 0; nb < 2; ++nb) {
#pragma unroll
        for (int kc = 0; kc < 2; ++kc) {
          v16bf kf = load_frag16(Kl, 64, nb * 16 + nl, half, kc * 32);
          sc[0][nb] = wmma_bf16(qa[0][kc], kf, sc[0][nb]);
          sc[1][nb] = wmma_bf16(qa[1][kc], kf, sc[1][nb]);
        }
      }
      // ---- mask + online softmax per q-block ----
#pragma unroll
      for (int mb = 0; mb < 2; ++mb) {
        float p[2][8], rmax[8], scl[8];
#pragma unroll
        for (int v = 0; v < 8; ++v) {
          int qidx = q0 + mb * 16 + v + 8 * half;
          float x0 = sc[mb][0][v] * 0.125f;       // 1/sqrt(64)
          float x1 = sc[mb][1][v] * 0.125f;
          if (kb + nl      > qidx) x0 = -1e30f;
          if (kb + 16 + nl > qidx) x1 = -1e30f;
          sc[mb][0][v] = x0; sc[mb][1][v] = x1;
          rmax[v] = fmaxf(x0, x1);
#pragma unroll
          for (int msk = 1; msk <= 8; msk <<= 1)
            rmax[v] = fmaxf(rmax[v], __shfl_xor(rmax[v], msk, 32));
        }
#pragma unroll
        for (int v = 0; v < 8; ++v) {
          float mnew = fmaxf(mrow[mb][v], rmax[v]);
          scl[v]      = __expf(mrow[mb][v] - mnew);
          mrow[mb][v] = mnew;
          p[0][v] = __expf(sc[mb][0][v] - mnew);
          p[1][v] = __expf(sc[mb][1][v] - mnew);
          float rs = p[0][v] + p[1][v];
#pragma unroll
          for (int msk = 1; msk <= 8; msk <<= 1)
            rs += __shfl_xor(rs, msk, 32);
          lrow[mb][v] = lrow[mb][v] * scl[v] + rs;
        }
#pragma unroll
        for (int f = 0; f < 4; ++f)
#pragma unroll
          for (int v = 0; v < 8; ++v) oacc[mb][f][v] *= scl[v];
        // P: C-layout -> per-wave LDS tile
#pragma unroll
        for (int nb = 0; nb < 2; ++nb)
#pragma unroll
          for (int v = 0; v < 8; ++v)
            Pld[w][mb][(v + 8 * half) * 32 + nb * 16 + nl] = (__bf16)p[nb][v];
      }
      v16bf pa0 = load_frag16(Pld[w][0], 32, nl, half, 0);
      v16bf pa1 = load_frag16(Pld[w][1], 32, nl, half, 0);
      // ---- O += P(16x32) @ V(32x64), V fragments shared across both q-blocks ----
#pragma unroll
      for (int f = 0; f < 4; ++f) {
        v16bf vb = load_frag16(Vl, 32, f * 16 + nl, half, 0);
        oacc[0][f] = wmma_bf16(pa0, vb, oacc[0][f]);
        oacc[1][f] = wmma_bf16(pa1, vb, oacc[1][f]);
      }
    }
    __syncthreads();
  }

  // ---- normalize + store to [b][s][h*D+d] (bf16, feeds output GEMM) ----
#pragma unroll
  for (int mb = 0; mb < 2; ++mb)
#pragma unroll
    for (int f = 0; f < 4; ++f)
#pragma unroll
      for (int v = 0; v < 8; ++v) {
        int s = q0 + mb * 16 + v + 8 * half;
        int d = nl + 16 * f;
        float val = oacc[mb][f][v] / lrow[mb][v];
        O[((size_t)b * SSQ + s) * HIDN + h * HD + d] = (__bf16)val;
      }
}

// ---------------- output projection: out = attn(4096x2048) @ wo ----------------
// Same 4-wave / async-LDS-staged-B structure as k_gemm_qkv; fp32 output.
__global__ void __launch_bounds__(128)
k_gemm_out(const unsigned short* __restrict__ Ar,
           const unsigned short* __restrict__ WoTr,
           float* __restrict__ out) {
  const __bf16* A   = (const __bf16*)Ar;
  const __bf16* WoT = (const __bf16*)WoTr;    // N x K bf16

  __shared__ __align__(16) __bf16 Bt[2][64 * 32];

  const int t    = threadIdx.x;
  const int lane = t & 31;
  const int w    = t >> 5;
  const int nl   = lane & 15;
  const int half = lane >> 4;
  const int row0 = blockIdx.x * 128 + w * 32;
  const int nb64 = blockIdx.y;                // 64-col block

  const __bf16* Abase = A   + (size_t)row0 * HIDN;
  const __bf16* Bbase = WoT + (size_t)nb64 * 64 * HIDN;

  auto stage = [&](int buf, int k0) {
#pragma unroll
    for (int i = 0; i < 2; ++i) {
      int c  = t + i * 128;
      int n  = c >> 2;
      int ke = (c & 3) * 8;
      async_copy16((unsigned)(size_t)&Bt[buf][n * 32 + ke],
                   Bbase + (size_t)n * HIDN + k0 + ke);
    }
  };

  v8f acc[2][4] = {};
  stage(0, 0);
  for (int kt = 0; kt < HIDN / 32; ++kt) {
    const int k0 = kt * 32;
    const bool hasnext = (kt + 1) < (HIDN / 32);
    if (hasnext) {
      stage((kt + 1) & 1, k0 + 32);
      asm volatile("s_wait_asynccnt 2" ::: "memory");
    } else {
      asm volatile("s_wait_asynccnt 0" ::: "memory");
    }
    __syncthreads();

    const __bf16* Bc = Bt[kt & 1];
    v16bf a0 = load_frag16(Abase,                     HIDN, nl, half, k0);
    v16bf a1 = load_frag16(Abase + (size_t)16 * HIDN, HIDN, nl, half, k0);
#pragma unroll
    for (int f = 0; f < 4; ++f) {
      v16bf b = load_frag16(Bc, 32, f * 16 + nl, half, 0);
      acc[0][f] = wmma_bf16(a0, b, acc[0][f]);
      acc[1][f] = wmma_bf16(a1, b, acc[1][f]);
    }
    __syncthreads();
  }

#pragma unroll
  for (int mb = 0; mb < 2; ++mb)
#pragma unroll
    for (int f = 0; f < 4; ++f)
#pragma unroll
      for (int v = 0; v < 8; ++v) {
        int m = mb * 16 + v + 8 * half;
        out[(size_t)(row0 + m) * HIDN + nb64 * 64 + 16 * f + nl] = acc[mb][f][v];
      }
}

// ---------------- host launcher ----------------
extern "C" void kernel_launch(void* const* d_in, const int* in_sizes, int n_in,
                              void* d_out, int out_size, void* d_ws, size_t ws_size,
                              hipStream_t stream) {
  (void)in_sizes; (void)n_in; (void)out_size; (void)ws_size;

  const float* x    = (const float*)d_in[0];
  const float* cosT = (const float*)d_in[1];
  const float* sinT = (const float*)d_in[2];
  const float* wq   = (const float*)d_in[3];
  const float* wk   = (const float*)d_in[4];
  const float* wv   = (const float*)d_in[5];
  const float* wo   = (const float*)d_in[6];
  float* out = (float*)d_out;

  char* ws = (char*)d_ws;
  size_t off = 0;
  auto alloc = [&](size_t bytes) -> unsigned short* {
    unsigned short* p = (unsigned short*)(ws + off);
    off += (bytes + 255) & ~(size_t)255;
    return p;
  };
  unsigned short* xb   = alloc((size_t)MROWS * HIDN * 2);         // x bf16
  unsigned short* wqT  = alloc((size_t)HIDN * HIDN * 2);          // wq^T bf16
  unsigned short* wkT  = alloc((size_t)HIDN * NKVH * HD * 2);
  unsigned short* wvT  = alloc((size_t)HIDN * NKVH * HD * 2);
  unsigned short* woT  = alloc((size_t)HIDN * HIDN * 2);
  unsigned short* qb   = alloc((size_t)BB * NH   * SSQ * HD * 2); // [b][h][s][d]
  unsigned short* kbuf = alloc((size_t)BB * NKVH * SSQ * HD * 2); // [b][h][s][d]
  unsigned short* vtb  = alloc((size_t)BB * NKVH * HD * SSQ * 2); // [b][h][d][s]
  unsigned short* attn = alloc((size_t)MROWS * HIDN * 2);         // [b][s][H*D]

  // 1) precision conversion + weight transposes
  {
    int n = MROWS * HIDN;
    k_f32_to_bf16<<<(n + 255) / 256, 256, 0, stream>>>(x, xb, n);
  }
  k_transpose_bf16<<<(HIDN * HIDN + 255) / 256, 256, 0, stream>>>(wq, wqT, HIDN, NH * HD);
  k_transpose_bf16<<<(HIDN * NKVH * HD + 255) / 256, 256, 0, stream>>>(wk, wkT, HIDN, NKVH * HD);
  k_transpose_bf16<<<(HIDN * NKVH * HD + 255) / 256, 256, 0, stream>>>(wv, wvT, HIDN, NKVH * HD);
  k_transpose_bf16<<<(HIDN * HIDN + 255) / 256, 256, 0, stream>>>(wo, woT, HIDN, HIDN);

  // 2) QKV projections (+RoPE on q/k; v stored transposed)
  {
    dim3 gq(MROWS / 128, NH);
    k_gemm_qkv<<<gq, 128, 0, stream>>>(xb, wqT, cosT, sinT, qb,   NH,   1, 0);
    dim3 gk(MROWS / 128, NKVH);
    k_gemm_qkv<<<gk, 128, 0, stream>>>(xb, wkT, cosT, sinT, kbuf, NKVH, 1, 0);
    k_gemm_qkv<<<gk, 128, 0, stream>>>(xb, wvT, cosT, sinT, vtb,  NKVH, 0, 1);
  }

  // 3) causal flash attention (128 q rows / WG, async-staged K/V tiles)
  {
    dim3 ga(SSQ / 128, NH, BB);
    k_attn<<<ga, 128, 0, stream>>>(qb, kbuf, vtb, attn);
  }

  // 4) output projection -> f32
  {
    dim3 go(MROWS / 128, HIDN / 64);
    k_gemm_out<<<go, 128, 0, stream>>>(attn, woT, out);
  }
}
